// HybridGraphConstructor_87634512708197
// MI455X (gfx1250) — matrix operand order repaired and checked by reading, not compiled, not run on hardware
//
#include <hip/hip_runtime.h>
#include <math.h>

// ---------------- CDNA5 WMMA types ----------------
typedef __attribute__((ext_vector_type(16))) _Float16 v16h;
typedef __attribute__((ext_vector_type(8)))  _Float16 v8h;
typedef __attribute__((ext_vector_type(8)))  float    v8f;

// Problem constants
constexpr int kB  = 128;   // batch
constexpr int kN  = 22;    // nodes
constexpr int kF  = 448;   // features
constexpr int kH  = 64;    // hidden
constexpr int kMP = 32;    // nodes padded to 2x16 for WMMA M tiles
constexpr int kPC = 256;   // projection columns: [A1|A2|X1|X2] = 4*64

__device__ __forceinline__ float wave_sum(float v) {
#pragma unroll
  for (int m = 16; m >= 1; m >>= 1) v += __shfl_xor(v, m, 32);
  return v;
}
__device__ __forceinline__ float sigmoidf(float x) { return 1.0f / (1.0f + __expf(-x)); }

// Build a v16h A/B fragment from two 16-byte contiguous chunks.
__device__ __forceinline__ v16h load_frag(const _Float16* p0, const _Float16* p1) {
  v8h lo = *(const v8h*)p0;
  v8h hi = *(const v8h*)p1;
  v16h r;
#pragma unroll
  for (int t = 0; t < 8; ++t) { r[t] = lo[t]; r[t + 8] = hi[t]; }
  return r;
}

// ---------------------------------------------------------------------------
// K0a: pack [cw1_lo | cw1_hi | sw1_x1 | sw1_x2] as f16, transposed (col-major
// over K) so B fragments load K-contiguous.  WcatT[n][k], n<256, k<448.
// ---------------------------------------------------------------------------
__global__ void k_pack_weights(const float* __restrict__ cw1,
                               const float* __restrict__ sw1,
                               _Float16* __restrict__ wcatT) {
  int idx = blockIdx.x * blockDim.x + threadIdx.x;
  if (idx >= 256 * kF) return;
  int n = idx / kF, k = idx % kF;
  float v;
  if (n < 64)       v = cw1[(size_t)k * 64 + n];                  // x_i part of cw1
  else if (n < 128) v = cw1[(size_t)(kF + k) * 64 + (n - 64)];    // x_j part of cw1
  else if (n < 192) v = sw1[(size_t)(2 * kH + k) * 64 + (n - 128)];        // x_i part of sw1
  else              v = sw1[(size_t)(2 * kH + kF + k) * 64 + (n - 192)];   // x_j part of sw1
  wcatT[(size_t)n * kF + k] = (_Float16)v;
}

// ---------------------------------------------------------------------------
// K0b: E1 = emb @ sw1[0:64], E2 = emb @ sw1[64:128]   (tiny: 22x64 each)
// ---------------------------------------------------------------------------
__global__ void k_emb_proj(const float* __restrict__ emb,
                           const float* __restrict__ sw1,
                           float* __restrict__ E1, float* __restrict__ E2) {
  int idx = blockIdx.x * blockDim.x + threadIdx.x;
  if (idx >= kN * kH) return;
  int i = idx / kH, n = idx % kH;
  float a = 0.f, b = 0.f;
#pragma unroll 8
  for (int k = 0; k < kH; ++k) {
    float e = emb[i * kH + k];
    a = fmaf(e, sw1[(size_t)k * kH + n], a);
    b = fmaf(e, sw1[(size_t)(kH + k) * kH + n], b);
  }
  E1[idx] = a; E2[idx] = b;
}

// ---------------------------------------------------------------------------
// K1: per-row mean / unbiased std; pack x and xn=(x-mu)/(std+1e-8) to f16,
// rows padded to 32 per batch (pad rows = 0).  One wave per padded row.
// ---------------------------------------------------------------------------
__global__ void k_row_stats_pack(const float* __restrict__ x,
                                 _Float16* __restrict__ xh,
                                 _Float16* __restrict__ xnh) {
  int wave = (blockIdx.x * blockDim.x + threadIdx.x) >> 5;
  int lane = threadIdx.x & 31;
  if (wave >= kB * kMP) return;
  int b = wave / kMP, mi = wave % kMP;
  _Float16* xr  = xh  + (size_t)wave * kF;
  _Float16* xnr = xnh + (size_t)wave * kF;
  if (mi >= kN) {  // zero pad rows
    for (int k = lane; k < kF; k += 32) { xr[k] = (_Float16)0.f; xnr[k] = (_Float16)0.f; }
    return;
  }
  const float* row = x + ((size_t)b * kN + mi) * kF;
  float vals[kF / 32];
  float s = 0.f, ss = 0.f;
#pragma unroll
  for (int t = 0; t < kF / 32; ++t) {
    float v = row[lane + 32 * t];
    vals[t] = v; s += v; ss = fmaf(v, v, ss);
  }
  s = wave_sum(s); ss = wave_sum(ss);
  float mu  = s * (1.0f / kF);
  float var = (ss - (float)kF * mu * mu) * (1.0f / (kF - 1));   // torch ddof=1
  var = var < 0.f ? 0.f : var;
  float inv = 1.0f / (sqrtf(var) + 1e-8f);
#pragma unroll
  for (int t = 0; t < kF / 32; ++t) {
    int k = lane + 32 * t;
    xr[k]  = (_Float16)vals[t];
    xnr[k] = (_Float16)((vals[t] - mu) * inv);
  }
}

// ---------------------------------------------------------------------------
// K2: WMMA node projections  P[b] (32 x 256) = xh[b] (32x448) @ WcatT^T.
// Block = 16 waves; block per (batch, mtile); wave w owns 16-col N-tile w.
// ---------------------------------------------------------------------------
__global__ void k_proj_wmma(const _Float16* __restrict__ xh,
                            const _Float16* __restrict__ wcatT,
                            float* __restrict__ P) {
  int b     = blockIdx.x >> 1;
  int mtile = blockIdx.x & 1;
  int wave  = threadIdx.x >> 5;   // 0..15 -> N tile
  int lane  = threadIdx.x & 31;
  int half  = (lane < 16) ? 0 : 1;
  int l15   = lane & 15;

  // A: row M = mtile*16 + l15 ; lane<16 holds K {0..7,16..23}, lane>=16 {8..15,24..31}
  const _Float16* aptr = xh + ((size_t)b * kMP + mtile * 16 + l15) * kF + half * 8;
  // B: column n = wave*16 + l15 ; lane<16 holds K {0..7,16..23}, lane>=16 {8..15,24..31}
  const _Float16* bptr = wcatT + (size_t)(wave * 16 + l15) * kF + half * 8;

  v8f acc = {};
  for (int kk = 0; kk < kF; kk += 32) {
    v16h a = load_frag(aptr + kk, aptr + kk + 16);
    v16h bb = load_frag(bptr + kk, bptr + kk + 16);
    acc = __builtin_amdgcn_wmma_f32_16x16x32_f16(false, a, false, bb,
                                                 (short)0, acc, false, false);
  }
  // D layout: VGPR r -> M = r (+8 for upper half-lanes), N = l15
  float* prow = P + (size_t)b * kMP * kPC;
  int c = wave * 16 + l15;
  int rbase = mtile * 16 + half * 8;
#pragma unroll
  for (int r = 0; r < 8; ++r) prow[(size_t)(rbase + r) * kPC + c] = acc[r];
}

// ---------------------------------------------------------------------------
// K3: WMMA correlation  corr[b] = |xn @ xn^T| / F.  Block=4 waves per batch,
// wave owns one 16x16 (itile,jtile) output tile.
// ---------------------------------------------------------------------------
__global__ void k_corr_wmma(const _Float16* __restrict__ xnh,
                            float* __restrict__ corr) {
  int b    = blockIdx.x;
  int wave = threadIdx.x >> 5;   // 0..3
  int lane = threadIdx.x & 31;
  int itile = wave >> 1, jtile = wave & 1;
  int half = (lane < 16) ? 0 : 1;
  int l15  = lane & 15;
  const _Float16* base = xnh + (size_t)b * kMP * kF;
  const _Float16* aptr = base + (size_t)(itile * 16 + l15) * kF + half * 8;
  // B = xn^T: column n is node j -> K-contiguous along that node's row
  const _Float16* bptr = base + (size_t)(jtile * 16 + l15) * kF + half * 8;

  v8f acc = {};
  for (int kk = 0; kk < kF; kk += 32) {
    v16h a = load_frag(aptr + kk, aptr + kk + 16);
    v16h bb = load_frag(bptr + kk, bptr + kk + 16);
    acc = __builtin_amdgcn_wmma_f32_16x16x32_f16(false, a, false, bb,
                                                 (short)0, acc, false, false);
  }
  int j = jtile * 16 + l15;
  int rbase = itile * 16 + half * 8;
#pragma unroll
  for (int r = 0; r < 8; ++r) {
    int i = rbase + r;
    if (i < kN && j < kN)
      corr[((size_t)b * kN + i) * kN + j] = fabsf(acc[r] * (1.0f / kF));
  }
}

// ---------------------------------------------------------------------------
// K4: per-pair tail.  One wave per ordered pair (b,i,j).
// ---------------------------------------------------------------------------
__global__ void k_pairs(const float* __restrict__ P, const float* __restrict__ corrM,
                        const float* __restrict__ E1, const float* __restrict__ E2,
                        const float* __restrict__ cb1, const float* __restrict__ cw2,
                        const float* __restrict__ cb2, const float* __restrict__ sb1,
                        const float* __restrict__ lng, const float* __restrict__ lnb,
                        const float* __restrict__ sw2, const float* __restrict__ sb2,
                        const float* __restrict__ sw3, const float* __restrict__ sb3,
                        const float* __restrict__ thr, const float* __restrict__ alpha_p,
                        float* __restrict__ out) {
  __shared__ float s_sw2[kH * 32];
  __shared__ float s_cb1[kH], s_cw2[kH], s_sb1[kH], s_lng[kH], s_lnb[kH];
  __shared__ float s_sb2[32], s_sw3[32];
  int tid = threadIdx.x;
  for (int k = tid; k < kH * 32; k += blockDim.x) s_sw2[k] = sw2[k];
  if (tid < kH) { s_cb1[tid] = cb1[tid]; s_cw2[tid] = cw2[tid];
                  s_sb1[tid] = sb1[tid]; s_lng[tid] = lng[tid]; s_lnb[tid] = lnb[tid]; }
  if (tid < 32) { s_sb2[tid] = sb2[tid]; s_sw3[tid] = sw3[tid]; }
  __syncthreads();

  int wv = tid >> 5, lane = tid & 31;
  long p = (long)blockIdx.x * (blockDim.x / 32) + wv;
  if (p >= (long)kB * kN * kN) return;
  int b = (int)(p / (kN * kN));
  int r = (int)(p % (kN * kN));
  int i = r / kN, j = r % kN;
  const float* Pb = P + (size_t)b * kMP * kPC;
  int l0 = lane, l1 = lane + 32;

  // ---- correlation branch: h=relu(A1[i]+A2[j]+cb1); w=sigmoid(h.cw2+cb2)
  float h0 = fmaxf(Pb[i * kPC + l0] + Pb[j * kPC + 64 + l0] + s_cb1[l0], 0.f);
  float h1 = fmaxf(Pb[i * kPC + l1] + Pb[j * kPC + 64 + l1] + s_cb1[l1], 0.f);
  float wsum = wave_sum(h0 * s_cw2[l0] + h1 * s_cw2[l1]);
  float w_corr = sigmoidf(wsum + cb2[0]);
  float g_corr = corrM[((size_t)b * kN + i) * kN + j] * w_corr + (i == j ? 1.0f : 0.0f);

  // ---- semantic branch: symmetric via (min,max); diagonal stays identity
  float g_sem;
  if (i == j) {
    g_sem = 1.0f;
  } else {
    int ia = i < j ? i : j;
    int ja = i < j ? j : i;
    float s0 = E1[ia * kH + l0] + E2[ja * kH + l0]
             + Pb[ia * kPC + 128 + l0] + Pb[ja * kPC + 192 + l0] + s_sb1[l0];
    float s1 = E1[ia * kH + l1] + E2[ja * kH + l1]
             + Pb[ia * kPC + 128 + l1] + Pb[ja * kPC + 192 + l1] + s_sb1[l1];
    float mu  = wave_sum(s0 + s1) * (1.0f / kH);
    float d0 = s0 - mu, d1 = s1 - mu;
    float var = wave_sum(d0 * d0 + d1 * d1) * (1.0f / kH);
    float rstd = rsqrtf(var + 1e-5f);
    float h1a = fmaxf(fmaf(d0 * rstd, s_lng[l0], s_lnb[l0]), 0.f);
    float h1b = fmaxf(fmaf(d1 * rstd, s_lng[l1], s_lnb[l1]), 0.f);
    // h2 = relu(h1 @ sw2 + sb2): broadcast h1[k] via shfl, sw2 from LDS
    float acc2 = 0.f;
#pragma unroll 8
    for (int k = 0; k < 32; ++k) {
      acc2 = fmaf(__shfl(h1a, k, 32), s_sw2[k * 32 + lane], acc2);
      acc2 = fmaf(__shfl(h1b, k, 32), s_sw2[(k + 32) * 32 + lane], acc2);
    }
    float h2 = fmaxf(acc2 + s_sb2[lane], 0.f);
    float w_sem = sigmoidf(wave_sum(h2 * s_sw3[lane]) + sb3[0]);
    float t = sigmoidf(thr[0]);
    g_sem = (w_sem > t) ? w_sem : 0.0f;
  }

  float alpha = sigmoidf(alpha_p[0]);
  if (lane == 0)
    out[((size_t)b * kN + i) * kN + j] = alpha * g_corr + (1.0f - alpha) * g_sem;
}

// ---------------------------------------------------------------------------
extern "C" void kernel_launch(void* const* d_in, const int* in_sizes, int n_in,
                              void* d_out, int out_size, void* d_ws, size_t ws_size,
                              hipStream_t stream) {
  const float* x    = (const float*)d_in[0];
  const float* cw1  = (const float*)d_in[1];
  const float* cb1  = (const float*)d_in[2];
  const float* cw2  = (const float*)d_in[3];
  const float* cb2  = (const float*)d_in[4];
  const float* emb  = (const float*)d_in[5];
  const float* sw1  = (const float*)d_in[6];
  const float* sb1  = (const float*)d_in[7];
  const float* ln_g = (const float*)d_in[8];
  const float* ln_b = (const float*)d_in[9];
  const float* sw2  = (const float*)d_in[10];
  const float* sb2  = (const float*)d_in[11];
  const float* sw3  = (const float*)d_in[12];
  const float* sb3  = (const float*)d_in[13];
  const float* thr  = (const float*)d_in[14];
  const float* alph = (const float*)d_in[15];
  float* out = (float*)d_out;

  // workspace layout (256B-aligned offsets)
  char* ws = (char*)d_ws;
  _Float16* wcatT = (_Float16*)(ws);                       // 256*448*2   = 229376
  _Float16* xh    = (_Float16*)(ws + 229376);              // 128*32*448*2= 3670016
  _Float16* xnh   = (_Float16*)(ws + 3899392);             // same
  float*    P     = (float*)   (ws + 7569408);             // 128*32*256*4= 4194304
  float*    corr  = (float*)   (ws + 11763712);            // 128*22*22*4 = 247808
  float*    E1    = (float*)   (ws + 12011520);            // 22*64*4     = 5632
  float*    E2    = (float*)   (ws + 12017152);            // total ~12 MB

  // 1. weight pack (f16, transposed) + emb projections
  k_pack_weights<<<(256 * kF + 255) / 256, 256, 0, stream>>>(cw1, sw1, wcatT);
  k_emb_proj<<<(kN * kH + 255) / 256, 256, 0, stream>>>(emb, sw1, E1, E2);

  // 2. per-row stats + f16 packing (one wave per padded row; 4096 waves)
  k_row_stats_pack<<<(kB * kMP * 32 + 255) / 256, 256, 0, stream>>>(x, xh, xnh);

  // 3. WMMA node projections: grid (b, mtile), 16 waves/block
  k_proj_wmma<<<kB * 2, 512, 0, stream>>>(xh, wcatT, P);

  // 4. WMMA correlation matrices: 4 waves/block, block per batch
  k_corr_wmma<<<kB, 128, 0, stream>>>(xnh, corr);

  // 5. per-pair tail: wave per ordered pair, 8 waves/block
  long pairs = (long)kB * kN * kN;
  int blocks = (int)((pairs + 7) / 8);
  k_pairs<<<blocks, 256, 0, stream>>>(P, corr, E1, E2, cb1, cw2, cb2, sb1,
                                      ln_g, ln_b, sw2, sb2, sw3, sb3,
                                      thr, alph, out);
}
